// RetinaNet_24335284699340
// MI455X (gfx1250) — matrix operand order repaired and checked by reading, not compiled, not run on hardware
//
#include <hip/hip_runtime.h>
#include <hip/hip_bf16.h>

typedef __attribute__((ext_vector_type(16))) __bf16 v16bf;
typedef __attribute__((ext_vector_type(8)))  __bf16 v8bf;
typedef __attribute__((ext_vector_type(8)))  float  v8f;

// ---------------- zero fill (16B granules) ----------------
__global__ void k_zero(uint4* __restrict__ p, long n16) {
  long idx = (long)blockIdx.x * blockDim.x + threadIdx.x;
  if (idx < n16) p[idx] = make_uint4(0u, 0u, 0u, 0u);
}

// ------- convert NCDHW f32 -> spatially padded channels-last bf16 -------
__global__ void k_to_cl_bf16(const float* __restrict__ in, __bf16* __restrict__ out,
                             int C, int D, int H, int W) {
  int DHW = D * H * W;
  int idx = blockIdx.x * blockDim.x + threadIdx.x;
  if (idx >= C * DHW) return;
  int c = idx / DHW, v = idx - c * DHW;
  int w = v % W, h = (v / W) % H, d = v / (W * H);
  int Hp = H + 2, Wp = W + 2;
  int vp = ((d + 1) * Hp + (h + 1)) * Wp + (w + 1);
  out[(size_t)vp * C + c] = (__bf16)in[idx];
}

// ------- pack conv weights [Cout,Cin,3,3,3] f32 -> WMMA A-fragment bf16 -------
// layout: [coTile][tap][ciChunk][lane 0..31][e 0..15], K = g*16 + e
__global__ void k_pack_w(const float* __restrict__ w, __bf16* __restrict__ out,
                         int Cin, int Cout, int nCoTiles, int nChunks) {
  int idx = blockIdx.x * blockDim.x + threadIdx.x;
  int total = nCoTiles * 27 * nChunks * 512;
  if (idx >= total) return;
  int e    = idx & 15;
  int lane = (idx >> 4) & 31;
  int c    = (idx >> 9) % nChunks;
  int tap  = (idx / (512 * nChunks)) % 27;
  int t    = idx / (512 * nChunks * 27);
  int M = lane & 15, g = lane >> 4;
  int K = (g << 4) + e;
  int co = t * 16 + M, ci = c * 32 + K;
  float v = (co < Cout) ? w[((size_t)co * Cin + ci) * 27 + tap] : 0.0f;
  out[idx] = (__bf16)v;
}

// ---- nearest x2 upsample (padded coarse) + add (padded fine) -> padded fine ----
__global__ void k_upadd(const __bf16* __restrict__ src, const __bf16* __restrict__ add,
                        __bf16* __restrict__ out, int C, int D, int H, int W) {
  int idx = blockIdx.x * blockDim.x + threadIdx.x;
  int total = D * H * W * C;
  if (idx >= total) return;
  int c = idx % C, v = idx / C;
  int w = v % W, h = (v / W) % H, d = v / (W * H);
  int Hp = H + 2, Wp = W + 2;
  int Hsp = (H >> 1) + 2, Wsp = (W >> 1) + 2;
  int vp  = ((d + 1) * Hp + (h + 1)) * Wp + (w + 1);
  int vsp = (((d >> 1) + 1) * Hsp + ((h >> 1) + 1)) * Wsp + ((w >> 1) + 1);
  out[(size_t)vp * C + c] =
      (__bf16)((float)src[(size_t)vsp * C + c] + (float)add[(size_t)vp * C + c]);
}

// -------- scatter padded-channel f32 head output into d_out --------
__global__ void k_scatter(const float* __restrict__ buf, float* __restrict__ out,
                          int DHW, int CoutPad, int AK, size_t base) {
  int idx = blockIdx.x * blockDim.x + threadIdx.x;
  if (idx >= DHW * AK) return;
  int ck = idx % AK, v = idx / AK;
  out[base + (size_t)idx] = buf[(size_t)v * CoutPad + ck];
}

// ---------------- implicit-GEMM 3x3x3 conv via bf16 WMMA ----------------
// x: padded channels-last bf16 [(D+2)(H+2)(W+2), Cin] with zero halo.
// Wave computes a 16-outch x (VT*16)-voxel tile. Inner loop is software-
// pipelined with two register fragment buffers so each WMMA group consumes
// loads issued a full iteration earlier (partial s_wait_loadcnt, not 0).
template <int NCH, int VT>   // NCH = Cin/32 (2,4,8)
__global__ __launch_bounds__(256) void k_conv3d_wmma(
    const __bf16* __restrict__ x, const __bf16* __restrict__ wp,
    const float* __restrict__ bias, void* __restrict__ y,
    int outF32, int doRelu, int Cout, int CoutPad, int D, int H, int W) {
  const int Cin = NCH * 32;
  const int Hp = H + 2, Wp = W + 2;
  const int DHW = D * H * W;
  const int nCoTiles = CoutPad >> 4;
  const int nVG = (DHW + VT * 16 - 1) / (VT * 16);
  int wid = blockIdx.x * (blockDim.x >> 5) + (threadIdx.x >> 5);
  if (wid >= nCoTiles * nVG) return;            // wave-uniform exit
  int coTile = wid % nCoTiles;
  int vGroup = wid / nCoTiles;

  int lane = threadIdx.x & 31;
  int n = lane & 15, g = lane >> 4;

  int vv[VT], vpad[VT];
  const __bf16* bptr[VT];
#pragma unroll
  for (int t = 0; t < VT; ++t) {
    int v = vGroup * (VT * 16) + t * 16 + n;
    vv[t] = v;
    int vc = (v < DHW) ? v : (DHW - 1);
    int d0 = vc / (H * W);
    int r0 = vc - d0 * H * W;
    int h0 = r0 / W;
    int w0 = r0 - h0 * W;
    int vp = ((d0 + 1) * Hp + (h0 + 1)) * Wp + (w0 + 1);
    vpad[t] = vp;
    bptr[t] = x + (size_t)vp * Cin + (g << 4);
  }

  v8f zacc = {};
  v8f acc[VT];
#pragma unroll
  for (int t = 0; t < VT; ++t) acc[t] = zacc;

  const __bf16* arow = wp + (size_t)coTile * 27 * NCH * 512 + lane * 16;

  const int TOT = 27 * NCH;                    // even for NCH in {2,4,8}

  auto loadFrag = [&](int i, v16bf& a, v16bf (&b)[VT]) {
    a = *(const v16bf*)(arow + (size_t)i * 512);
    int tap = i / NCH;                         // power-of-two divide
    int dd = tap / 9 - 1;
    int hh = (tap / 3) % 3 - 1;
    int ww = tap % 3 - 1;
    int toff = ((dd * Hp + hh) * Wp + ww) * Cin + (i % NCH) * 32;  // wave-uniform
#pragma unroll
    for (int t = 0; t < VT; ++t)
      b[t] = *(const v16bf*)(bptr[t] + toff);
  };
  auto wmmaGroup = [&](const v16bf& a, const v16bf (&b)[VT]) {
#pragma unroll
    for (int t = 0; t < VT; ++t)
      acc[t] = __builtin_amdgcn_wmma_f32_16x16x32_bf16(false, a, false, b[t],
                                                       (short)0, acc[t], false, false);
  };

  v16bf A0, A1, B0[VT], B1[VT];
  loadFrag(0, A0, B0);
  int i = 0;
  for (; i < TOT - 2; i += 2) {
    loadFrag(i + 1, A1, B1);
    wmmaGroup(A0, B0);                         // loads issued one iter earlier
    loadFrag(i + 2, A0, B0);
    wmmaGroup(A1, B1);
  }
  loadFrag(TOT - 1, A1, B1);                   // tail (i == TOT-2)
  wmmaGroup(A0, B0);
  wmmaGroup(A1, B1);

  int coBase = (coTile << 4) + (g << 3);
#pragma unroll
  for (int t = 0; t < VT; ++t) {
    if (vv[t] < DHW) {
      if (outF32) {
        float* o = (float*)y + (size_t)vv[t] * CoutPad + coBase;
#pragma unroll
        for (int r = 0; r < 8; ++r) {
          int co = coBase + r;
          float f = acc[t][r] + ((co < Cout) ? bias[co] : 0.0f);
          if (doRelu && f < 0.0f) f = 0.0f;
          o[r] = f;
        }
      } else {
        v8bf ob;
#pragma unroll
        for (int r = 0; r < 8; ++r) {
          int co = coBase + r;
          float f = acc[t][r] + ((co < Cout) ? bias[co] : 0.0f);
          if (doRelu && f < 0.0f) f = 0.0f;
          ob[r] = (__bf16)f;
        }
        *(v8bf*)((__bf16*)y + (size_t)vpad[t] * CoutPad + coBase) = ob;
      }
    }
  }
}

// ---------------- host-side helpers ----------------
static inline int cdiv(int a, int b) { return (a + b - 1) / b; }

static void conv_launch(hipStream_t s, const __bf16* x, const __bf16* wp,
                        const float* bias, void* y, int outF32, int relu,
                        int Cin, int Cout, int CoutPad, int D) {
  int DHW = D * D * D;
  int VT = (D >= 24) ? 4 : ((D >= 12) ? 2 : 1);
  int tiles = (CoutPad / 16) * cdiv(DHW, VT * 16);
  int blocks = cdiv(tiles, 8);   // 8 waves / block
#define CONV_GO(NCH_, VT_)                                                            \
  k_conv3d_wmma<NCH_, VT_><<<blocks, 256, 0, s>>>(x, wp, bias, y, outF32, relu,       \
                                                  Cout, CoutPad, D, D, D)
  if (Cin == 64) {
    if (VT == 4) CONV_GO(2, 4); else if (VT == 2) CONV_GO(2, 2); else CONV_GO(2, 1);
  } else if (Cin == 128) {
    if (VT == 4) CONV_GO(4, 4); else if (VT == 2) CONV_GO(4, 2); else CONV_GO(4, 1);
  } else {
    if (VT == 4) CONV_GO(8, 4); else if (VT == 2) CONV_GO(8, 2); else CONV_GO(8, 1);
  }
#undef CONV_GO
}

static void pack_launch(hipStream_t s, const float* w, __bf16* wp,
                        int Cin, int Cout, int CoutPad) {
  int nCo = CoutPad / 16, nCh = Cin / 32;
  int n = nCo * 27 * nCh * 512;
  k_pack_w<<<cdiv(n, 256), 256, 0, s>>>(w, wp, Cin, Cout, nCo, nCh);
}

extern "C" void kernel_launch(void* const* d_in, const int* in_sizes, int n_in,
                              void* d_out, int out_size, void* d_ws, size_t ws_size,
                              hipStream_t stream) {
  (void)in_sizes; (void)n_in; (void)out_size; (void)ws_size;
  const int D1 = 24, D2 = 12, D3 = 6;
  const int V1 = 13824, V2 = 1728, V3 = 216;
  const int PV1 = 26 * 26 * 26, PV2 = 14 * 14 * 14, PV3 = 8 * 8 * 8;  // padded vols

  const float* f1 = (const float*)d_in[0];
  const float* f2 = (const float*)d_in[1];
  const float* f3 = (const float*)d_in[2];
  const float* pt1w = (const float*)d_in[3];  const float* pt1b = (const float*)d_in[4];
  const float* pt2w = (const float*)d_in[5];  const float* pt2b = (const float*)d_in[6];
  const float* pt3w = (const float*)d_in[7];  const float* pt3b = (const float*)d_in[8];
  const float* ut1w = (const float*)d_in[9];  const float* ut1b = (const float*)d_in[10];
  const float* ut2w = (const float*)d_in[11]; const float* ut2b = (const float*)d_in[12];
  const float* bw[5], *bb[5], *cw[5], *cb[5];
  for (int i = 0; i < 5; ++i) {
    bw[i] = (const float*)d_in[13 + 2 * i]; bb[i] = (const float*)d_in[14 + 2 * i];
    cw[i] = (const float*)d_in[23 + 2 * i]; cb[i] = (const float*)d_in[24 + 2 * i];
  }
  float* out = (float*)d_out;

  // workspace layout
  char* ws = (char*)d_ws;
  size_t off = 0;
  auto alloc = [&](size_t b) { size_t o = off; off += (b + 255) & ~(size_t)255; return o; };
  __bf16* WP  = (__bf16*)(ws + alloc(4u << 20));              // packed weight slab
  size_t zStart = off;
  __bf16* F1C = (__bf16*)(ws + alloc((size_t)PV1 * 64 * 2));
  __bf16* F2C = (__bf16*)(ws + alloc((size_t)PV2 * 128 * 2));
  __bf16* F3C = (__bf16*)(ws + alloc((size_t)PV3 * 256 * 2));
  __bf16* P3  = (__bf16*)(ws + alloc((size_t)PV3 * 256 * 2));
  __bf16* P2  = (__bf16*)(ws + alloc((size_t)PV2 * 256 * 2));
  __bf16* P1  = (__bf16*)(ws + alloc((size_t)PV1 * 256 * 2));
  __bf16* T0  = (__bf16*)(ws + alloc((size_t)PV1 * 256 * 2));
  __bf16* T1  = (__bf16*)(ws + alloc((size_t)PV1 * 256 * 2));
  size_t zEnd = off;
  float* OUTF = (float*)(ws + alloc((size_t)V1 * 64 * 4));

  // ---- zero padded activation region (halos must be 0) ----
  long n16 = (long)((zEnd - zStart) / 16);
  k_zero<<<(int)((n16 + 255) / 256), 256, 0, stream>>>((uint4*)(ws + zStart), n16);

  // ---- activation converts (into padded interiors) ----
  k_to_cl_bf16<<<cdiv(64 * V1, 256), 256, 0, stream>>>(f1, F1C, 64, D1, D1, D1);
  k_to_cl_bf16<<<cdiv(128 * V2, 256), 256, 0, stream>>>(f2, F2C, 128, D2, D2, D2);
  k_to_cl_bf16<<<cdiv(256 * V3, 256), 256, 0, stream>>>(f3, F3C, 256, D3, D3, D3);

  // ---- FPN ----
  pack_launch(stream, pt3w, WP, 256, 256, 256);
  conv_launch(stream, F3C, WP, pt3b, P3, 0, 0, 256, 256, 256, D3);       // p3

  pack_launch(stream, pt2w, WP, 128, 256, 256);
  conv_launch(stream, F2C, WP, pt2b, T0, 0, 0, 128, 256, 256, D2);       // p2t
  k_upadd<<<cdiv(V2 * 256, 256), 256, 0, stream>>>(P3, T0, T1, 256, D2, D2, D2);
  pack_launch(stream, ut2w, WP, 256, 256, 256);
  conv_launch(stream, T1, WP, ut2b, P2, 0, 0, 256, 256, 256, D2);        // p2

  pack_launch(stream, pt1w, WP, 64, 256, 256);
  conv_launch(stream, F1C, WP, pt1b, T0, 0, 0, 64, 256, 256, D1);        // p1t
  k_upadd<<<cdiv(V1 * 256, 256), 256, 0, stream>>>(P2, T0, T1, 256, D1, D1, D1);
  pack_launch(stream, ut1w, WP, 256, 256, 256);
  conv_launch(stream, T1, WP, ut1b, P1, 0, 0, 256, 256, 256, D1);        // p1

  // ---- subnets: rows p1=0, p2=124416, p3=139968 ; boxes then classes ----
  const __bf16* feat[3] = {P1, P2, P3};
  const int     Dl[3]   = {D1, D2, D3};
  const size_t  boxBase[3] = {0, 746496, 839808};            // rowOff * 6
  const size_t  clsBase[3] = {851472, 1100304, 1131408};     // 851472 + rowOff * 2

  for (int L = 0; L < 3; ++L) {
    int D = Dl[L], DHW = D * D * D;
    for (int head = 0; head < 2; ++head) {
      const float* const* W = head ? cw : bw;
      const float* const* B = head ? cb : bb;
      int Cout = head ? 18 : 54;
      int CoutPad = head ? 32 : 64;
      size_t base = head ? clsBase[L] : boxBase[L];
      const __bf16* src = feat[L];
      __bf16* dsts[4] = {T0, T1, T0, T1};
      for (int i = 0; i < 4; ++i) {
        pack_launch(stream, W[i], WP, 256, 256, 256);
        conv_launch(stream, src, WP, B[i], dsts[i], 0, 1, 256, 256, 256, D);
        src = dsts[i];
      }
      pack_launch(stream, W[4], WP, 256, Cout, CoutPad);
      conv_launch(stream, src, WP, B[4], OUTF, 1, 0, 256, Cout, CoutPad, D);
      k_scatter<<<cdiv(DHW * Cout, 256), 256, 0, stream>>>(OUTF, out, DHW, CoutPad, Cout, base);
    }
  }
}
